// GRITAttention_84851373899972
// MI455X (gfx1250) — matrix operand order repaired and checked by reading, not compiled, not run on hardware
//
#include <hip/hip_runtime.h>
#include <hip/hip_bf16.h>

#define N_TOK   4096
#define DMODEL  512
#define D3      1536
#define HHEADS  8
#define HD      64
#define DPE     16
#define LOG2E   1.44269504088896340736f

typedef _Float16 v16h __attribute__((ext_vector_type(16)));
typedef _Float16 h8   __attribute__((ext_vector_type(8)));
typedef float    v8f  __attribute__((ext_vector_type(8)));

static __device__ __forceinline__ v16h combine16(h8 lo, h8 hi) {
    return __builtin_shufflevector(lo, hi, 0,1,2,3,4,5,6,7,8,9,10,11,12,13,14,15);
}

// CDNA5 async global->LDS copy (ASYNCcnt-tracked, no VGPR data return)
static __device__ __forceinline__ void async_copy_b128(void* lds_dst, const void* gsrc) {
    uint32_t l = (uint32_t)(uintptr_t)lds_dst;   // low 32 bits = LDS offset
    asm volatile("global_load_async_to_lds_b128 %0, %1, off"
                 :: "v"(l), "v"(gsrc) : "memory");
}
#define WAIT_ASYNC()  asm volatile("s_wait_asynccnt 0x0" ::: "memory")
#define WAIT_DS()     asm volatile("s_wait_dscnt 0x0"    ::: "memory")

// XOR-butterfly steps inside 8-lane groups via DPP8 (pure VALU, no LDS traffic)
template <int M>
static __device__ __forceinline__ float xor_max_dpp8(float v) {
    constexpr int S = ((0 ^ M)) | ((1 ^ M) << 3) | ((2 ^ M) << 6) | ((3 ^ M) << 9) |
                      ((4 ^ M) << 12) | ((5 ^ M) << 15) | ((6 ^ M) << 18) | ((7 ^ M) << 21);
    int o = __builtin_amdgcn_mov_dpp8(__builtin_bit_cast(int, v), S);
    return fmaxf(v, __builtin_bit_cast(float, o));
}
template <int M>
static __device__ __forceinline__ float xor_add_dpp8(float v) {
    constexpr int S = ((0 ^ M)) | ((1 ^ M) << 3) | ((2 ^ M) << 6) | ((3 ^ M) << 9) |
                      ((4 ^ M) << 12) | ((5 ^ M) << 15) | ((6 ^ M) << 18) | ((7 ^ M) << 21);
    int o = __builtin_amdgcn_mov_dpp8(__builtin_bit_cast(int, v), S);
    return v + __builtin_bit_cast(float, o);
}

// ---------------------------------------------------------------- converts
__global__ __launch_bounds__(256) void f32_to_f16_kernel(
    const float* __restrict__ src, _Float16* __restrict__ dst, int n) {
    int i = blockIdx.x * blockDim.x + threadIdx.x;
    int stride = gridDim.x * blockDim.x;
    for (; i < n; i += stride) dst[i] = (_Float16)src[i];
}

// --------------------------------- pe projection, pre-scaled by log2e [H][N]
__global__ __launch_bounds__(256) void peproj_kernel(
    const float* __restrict__ pe, const float* __restrict__ w_pe,
    const float* __restrict__ b_pe, float* __restrict__ peproj) {
    int id = blockIdx.x * blockDim.x + threadIdx.x;   // H*N threads
    int n = id & (N_TOK - 1);
    int h = id >> 12;                                 // N_TOK = 2^12
    float acc = b_pe[h];
#pragma unroll
    for (int d = 0; d < DPE; ++d)
        acc += pe[n * DPE + d] * w_pe[d * HHEADS + h];
    peproj[(size_t)h * N_TOK + n] = acc * LOG2E;      // softmax runs in log2 domain
}

// ---------------------------------------------------------------- GEMM
// MODE 0: C = A[4096xK] * B[KxD3] + bias  -> scatter f16 into head-major q/k/v
// MODE 1: C = A[4096xK] * B[KxD]  + bias  -> f32 to out
template <int MODE>
__global__ __launch_bounds__(256) void gemm_kernel(
    const _Float16* __restrict__ A, const _Float16* __restrict__ B,
    const float* __restrict__ bias,
    _Float16* __restrict__ qh, _Float16* __restrict__ kh, _Float16* __restrict__ vh,
    float* __restrict__ outf, int Kdim, int Ncols) {
    const int BM = 128, BN = 64, BK = 32, SA = 48, SB = 48;
    __shared__ __align__(16) _Float16 As[2][BM * SA];   // 2 x 12 KB, async-filled
    __shared__ __align__(16) _Float16 Bt[2][BN * SB];   // 2 x  6 KB, transposed

    const int tid = threadIdx.x;
    const int wave = tid >> 5, lane = tid & 31;
    const int lhalf = lane >> 4, lmod = lane & 15;
    const int row0 = blockIdx.y * BM;
    const int col0 = blockIdx.x * BN;

    // per-thread fixed source coordinates
    const int la0 = tid * 8,         ra0 = la0 >> 5, ca0 = la0 & 31;       // A chunk 0
    const int la1 = (256 + tid) * 8, ra1 = la1 >> 5, ca1 = la1 & 31;       // A chunk 1
    const int lb  = tid * 8,         kkb = lb >> 6,  nnb = lb & 63;        // B chunk

    v8f acc[4] = {};
    const int T = Kdim / BK;

    // prologue: issue tile 0
    async_copy_b128(&As[0][ra0 * SA + ca0], &A[(size_t)(row0 + ra0) * Kdim + ca0]);
    async_copy_b128(&As[0][ra1 * SA + ca1], &A[(size_t)(row0 + ra1) * Kdim + ca1]);
    h8 bb = *(const h8*)&B[(size_t)kkb * Ncols + col0 + nnb];

    for (int i = 0; i < T; ++i) {
        const int cur = i & 1, nxt = cur ^ 1;
        WAIT_ASYNC();                               // A(i) resident in LDS
#pragma unroll
        for (int u = 0; u < 8; ++u) Bt[cur][(nnb + u) * SB + kkb] = bb[u];
        __syncthreads();                            // tile i ready; compute i-1 done everywhere
        if (i + 1 < T) {
            const int k0 = (i + 1) * BK;
            async_copy_b128(&As[nxt][ra0 * SA + ca0],
                            &A[(size_t)(row0 + ra0) * Kdim + k0 + ca0]);
            async_copy_b128(&As[nxt][ra1 * SA + ca1],
                            &A[(size_t)(row0 + ra1) * Kdim + k0 + ca1]);
            bb = *(const h8*)&B[(size_t)(k0 + kkb) * Ncols + col0 + nnb];
        }
        const int m = wave * 16 + lmod;
        const int o1 = lhalf * 8;
        const int ob = lhalf * 8;
        v16h afrag = combine16(*(const h8*)&As[cur][m * SA + o1],
                               *(const h8*)&As[cur][m * SA + o1 + 16]);
#pragma unroll
        for (int t = 0; t < 4; ++t) {
            int n = t * 16 + lmod;
            v16h bfrag = combine16(*(const h8*)&Bt[cur][n * SB + ob],
                                   *(const h8*)&Bt[cur][n * SB + ob + 16]);
            acc[t] = __builtin_amdgcn_wmma_f32_16x16x32_f16(
                false, afrag, false, bfrag, (short)0, acc[t], false, false);
        }
    }

    if (MODE == 0) {
        int which = col0 / DMODEL;                 // 0=q 1=k 2=v
        int head = (col0 % DMODEL) / HD;           // col block == one head slice
        _Float16* dst = (which == 0) ? qh : (which == 1) ? kh : vh;
#pragma unroll
        for (int t = 0; t < 4; ++t) {
            int dcol = t * 16 + lmod;
            float bc = bias[col0 + dcol];
#pragma unroll
            for (int r = 0; r < 8; ++r) {
                int mm = row0 + wave * 16 + lhalf * 8 + r;
                dst[(size_t)head * N_TOK * HD + (size_t)mm * HD + dcol] =
                    (_Float16)(acc[t][r] + bc);
            }
        }
    } else {
#pragma unroll
        for (int t = 0; t < 4; ++t) {
            int cc = col0 + t * 16 + lmod;
            float bc = bias[cc];
#pragma unroll
            for (int r = 0; r < 8; ++r) {
                int mm = row0 + wave * 16 + lhalf * 8 + r;
                outf[(size_t)mm * DMODEL + cc] = acc[t][r] + bc;
            }
        }
    }
}

// ------------------------------------------------------ flash attention
__global__ __launch_bounds__(256) void attn_kernel(
    const _Float16* __restrict__ qh, const _Float16* __restrict__ kh,
    const _Float16* __restrict__ vh, const float* __restrict__ peproj,
    _Float16* __restrict__ aout) {
    const int BM = 128, BN = 64, SQ = 72, SK = 72, SV = 72;
    // QPs: first holds the Q tile; after Q fragments move to VGPRs each wave
    // reuses ITS OWN 16 rows as the P (C-layout -> A-layout) staging buffer.
    __shared__ __align__(16) _Float16 QPs[BM * SQ];       // 18 KB
    __shared__ __align__(16) _Float16 Ks[2][BN * SK];     // 2 x 9 KB (async)
    __shared__ __align__(16) _Float16 Vts[2][HD * SV];    // 2 x 9 KB (V^T)

    const int tid = threadIdx.x;
    const int wave = tid >> 5, lane = tid & 31;
    const int lhalf = lane >> 4, lmod = lane & 15;
    const int head = blockIdx.y;
    const int qb = blockIdx.x;
    const float scale = 0.125f * LOG2E;                   // hd^-0.5, log2 domain

    const _Float16* Qg  = qh + (size_t)head * N_TOK * HD + (size_t)qb * BM * HD;
    const _Float16* Kgh = kh + (size_t)head * N_TOK * HD;
    const _Float16* Vgh = vh + (size_t)head * N_TOK * HD;

    // per-thread fixed coordinates for 64x64 K/V tiles
    const int l0 = tid * 8,         r0 = l0 >> 6, c0 = l0 & 63;
    const int l1 = (256 + tid) * 8, r1 = l1 >> 6, c1 = l1 & 63;

    // ---- prologue: async Q tile + async K tile 0 + V tile 0 into regs
#pragma unroll
    for (int i = 0; i < 4; ++i) {
        int lin = (i * 256 + tid) * 8;
        int r = lin >> 6, c = lin & 63;
        async_copy_b128(&QPs[r * SQ + c], &Qg[(size_t)r * HD + c]);
    }
    async_copy_b128(&Ks[0][r0 * SK + c0], &Kgh[(size_t)r0 * HD + c0]);
    async_copy_b128(&Ks[0][r1 * SK + c1], &Kgh[(size_t)r1 * HD + c1]);
    h8 vv0 = *(const h8*)&Vgh[(size_t)r0 * HD + c0];
    h8 vv1 = *(const h8*)&Vgh[(size_t)r1 * HD + c1];
    WAIT_ASYNC();
    __syncthreads();

    // Q fragments -> registers (each wave reads only its own 16 rows)
    const int m = wave * 16 + lmod;
    const int o1 = lhalf * 8;
    v16h aq0 = combine16(*(const h8*)&QPs[m * SQ + o1],
                         *(const h8*)&QPs[m * SQ + o1 + 16]);
    v16h aq1 = combine16(*(const h8*)&QPs[m * SQ + 32 + o1],
                         *(const h8*)&QPs[m * SQ + 32 + o1 + 16]);
    _Float16* Pw = &QPs[wave * 16 * SQ];   // per-wave exclusive region

    float rmax[8], rsum[8];
    v8f co[4] = {};
#pragma unroll
    for (int r = 0; r < 8; ++r) { rmax[r] = -1e30f; rsum[r] = 0.0f; }
    const float* pj_base = peproj + (size_t)head * N_TOK;

    for (int kb = 0; kb < N_TOK / BN; ++kb) {
        const int cur = kb & 1, nxt = cur ^ 1;
        WAIT_ASYNC();                                  // K(kb) resident
        // V(kb): transposed scatter from regs (loads overlapped last iter)
#pragma unroll
        for (int u = 0; u < 8; ++u) Vts[cur][(c0 + u) * SV + r0] = vv0[u];
#pragma unroll
        for (int u = 0; u < 8; ++u) Vts[cur][(c1 + u) * SV + r1] = vv1[u];
        __syncthreads();                               // tile kb ready everywhere
        if (kb + 1 < N_TOK / BN) {                     // prefetch tile kb+1
            const _Float16* Kg = Kgh + (size_t)(kb + 1) * BN * HD;
            const _Float16* Vg = Vgh + (size_t)(kb + 1) * BN * HD;
            async_copy_b128(&Ks[nxt][r0 * SK + c0], &Kg[(size_t)r0 * HD + c0]);
            async_copy_b128(&Ks[nxt][r1 * SK + c1], &Kg[(size_t)r1 * HD + c1]);
            vv0 = *(const h8*)&Vg[(size_t)r0 * HD + c0];
            vv1 = *(const h8*)&Vg[(size_t)r1 * HD + c1];
        }

        // S = Q K^T   (K-tile rows are directly the B-fragment layout)
        v8f s[4] = {};
#pragma unroll
        for (int dc = 0; dc < HD; dc += 32) {
            v16h aq = (dc == 0) ? aq0 : aq1;
#pragma unroll
            for (int t = 0; t < 4; ++t) {
                int n = t * 16 + lmod;
                v16h bk = combine16(
                    *(const h8*)&Ks[cur][n * SK + dc + lhalf * 8],
                    *(const h8*)&Ks[cur][n * SK + dc + lhalf * 8 + 16]);
                s[t] = __builtin_amdgcn_wmma_f32_16x16x32_f16(
                    false, aq, false, bk, (short)0, s[t], false, false);
            }
        }

        // bias (row-constant part cancels in softmax) + online softmax (log2)
        float pj[4];
#pragma unroll
        for (int t = 0; t < 4; ++t) pj[t] = pj_base[kb * BN + t * 16 + lmod];

#pragma unroll
        for (int r = 0; r < 8; ++r) {
            float mx = rmax[r];
#pragma unroll
            for (int t = 0; t < 4; ++t) {
                float v = s[t][r] * scale + pj[t];
                s[t][r] = v;
                mx = fmaxf(mx, v);
            }
            mx = xor_max_dpp8<1>(mx);                  // VALU lane permutes
            mx = xor_max_dpp8<2>(mx);
            mx = xor_max_dpp8<4>(mx);
            mx = fmaxf(mx, __shfl_xor(mx, 8, 16));     // only cross-8 hop uses DS
            float alpha = exp2f(rmax[r] - mx);
            rmax[r] = mx;
            float ls = 0.0f;
#pragma unroll
            for (int t = 0; t < 4; ++t) {
                float p = exp2f(s[t][r] - mx);
                s[t][r] = p;
                ls += p;
            }
            ls = xor_add_dpp8<1>(ls);
            ls = xor_add_dpp8<2>(ls);
            ls = xor_add_dpp8<4>(ls);
            ls += __shfl_xor(ls, 8, 16);
            rsum[r] = rsum[r] * alpha + ls;
#pragma unroll
            for (int t = 0; t < 4; ++t) co[t][r] *= alpha;
        }

        // P: C-layout -> A-layout via per-wave LDS region (wave-private rows)
#pragma unroll
        for (int t = 0; t < 4; ++t)
#pragma unroll
            for (int r = 0; r < 8; ++r)
                Pw[(lhalf * 8 + r) * SQ + t * 16 + lmod] = (_Float16)s[t][r];
        WAIT_DS();

        // O += P V   (Vts is V^T so B-fragments are contiguous)
#pragma unroll
        for (int jc = 0; jc < BN; jc += 32) {
            v16h ap = combine16(*(const h8*)&Pw[lmod * SQ + jc + o1],
                                *(const h8*)&Pw[lmod * SQ + jc + o1 + 16]);
#pragma unroll
            for (int t = 0; t < 4; ++t) {
                int n = t * 16 + lmod;
                v16h bv = combine16(
                    *(const h8*)&Vts[cur][n * SV + jc + lhalf * 8],
                    *(const h8*)&Vts[cur][n * SV + jc + lhalf * 8 + 16]);
                co[t] = __builtin_amdgcn_wmma_f32_16x16x32_f16(
                    false, ap, false, bv, (short)0, co[t], false, false);
            }
        }
    }

    // epilogue: O / rowsum -> aout[n][h*64+d] (f16, feeds the out GEMM)
#pragma unroll
    for (int t = 0; t < 4; ++t) {
#pragma unroll
        for (int r = 0; r < 8; ++r) {
            int mm = qb * BM + wave * 16 + lhalf * 8 + r;
            aout[(size_t)mm * DMODEL + head * HD + t * 16 + lmod] =
                (_Float16)(co[t][r] / rsum[r]);
        }
    }
}

// ---------------------------------------------------------------- launch
extern "C" void kernel_launch(void* const* d_in, const int* in_sizes, int n_in,
                              void* d_out, int out_size, void* d_ws, size_t ws_size,
                              hipStream_t stream) {
    const float* x     = (const float*)d_in[0];
    const float* pe    = (const float*)d_in[1];
    const float* w_qkv = (const float*)d_in[2];
    const float* b_qkv = (const float*)d_in[3];
    const float* w_pe  = (const float*)d_in[4];
    const float* b_pe  = (const float*)d_in[5];
    const float* w_out = (const float*)d_in[6];
    const float* b_out = (const float*)d_in[7];
    float* out = (float*)d_out;

    char* ws = (char*)d_ws;
    size_t off = 0;
    auto alloc = [&](size_t bytes) {
        void* p = ws + off;
        off += (bytes + 255) & ~(size_t)255;
        return p;
    };
    _Float16* xh     = (_Float16*)alloc((size_t)N_TOK * DMODEL * 2);
    _Float16* wqkvh  = (_Float16*)alloc((size_t)DMODEL * D3 * 2);
    _Float16* wouth  = (_Float16*)alloc((size_t)DMODEL * DMODEL * 2);
    _Float16* qh     = (_Float16*)alloc((size_t)HHEADS * N_TOK * HD * 2);
    _Float16* kh     = (_Float16*)alloc((size_t)HHEADS * N_TOK * HD * 2);
    _Float16* vh     = (_Float16*)alloc((size_t)HHEADS * N_TOK * HD * 2);
    float*    pproj  = (float*)   alloc((size_t)HHEADS * N_TOK * 4);
    _Float16* aout   = (_Float16*)alloc((size_t)N_TOK * DMODEL * 2);

    f32_to_f16_kernel<<<1024, 256, 0, stream>>>(x, xh, N_TOK * DMODEL);
    f32_to_f16_kernel<<<512, 256, 0, stream>>>(w_qkv, wqkvh, DMODEL * D3);
    f32_to_f16_kernel<<<256, 256, 0, stream>>>(w_out, wouth, DMODEL * DMODEL);
    peproj_kernel<<<(HHEADS * N_TOK) / 256, 256, 0, stream>>>(pe, w_pe, b_pe, pproj);

    gemm_kernel<0><<<dim3(D3 / 64, N_TOK / 128), 256, 0, stream>>>(
        xh, wqkvh, b_qkv, qh, kh, vh, nullptr, DMODEL, D3);

    attn_kernel<<<dim3(N_TOK / 128, HHEADS), 256, 0, stream>>>(
        qh, kh, vh, pproj, aout);

    gemm_kernel<1><<<dim3(DMODEL / 64, N_TOK / 128), 256, 0, stream>>>(
        aout, wouth, b_out, nullptr, nullptr, nullptr, out, DMODEL, DMODEL);
}
